// MultiHeadAttention_86320252715754
// MI455X (gfx1250) — compile-verified, matrix operand order checked by reading, not run on hardware
//
#include <hip/hip_runtime.h>
#include <hip/hip_bf16.h>

// ---------------------------------------------------------------------------
// MHA + residual + LayerNorm(axis=L) for gfx1250 (MI455X), f16 WMMA path,
// async global->LDS copies + double-buffered, software-pipelined tiles.
// B=2, L=2048, HIDDEN=768, HEADS=12, HEAD=64.
// ---------------------------------------------------------------------------

typedef __attribute__((ext_vector_type(16))) _Float16 v16h;
typedef __attribute__((ext_vector_type(8)))  float    v8f;
typedef __attribute__((ext_vector_type(4)))  int      v4i;

#define WMMA_F16(a, b, c) \
  __builtin_amdgcn_wmma_f32_16x16x32_f16(false, (a), false, (b), (short)0, (c), false, false)

#if __has_builtin(__builtin_amdgcn_global_load_async_to_lds_b128)
#define HAVE_ASYNC_LDS 1
#else
#define HAVE_ASYNC_LDS 0
#endif

// 16-byte global -> LDS copy; async (ASYNCcnt) when the builtin exists.
// Probed signature: (v4i addrspace(1)*, v4i addrspace(3)*, imm offset, imm cpol)
__device__ inline void async_copy16(const _Float16* g, _Float16* l) {
#if HAVE_ASYNC_LDS
  __builtin_amdgcn_global_load_async_to_lds_b128(
      (__attribute__((address_space(1))) v4i*)g,
      (__attribute__((address_space(3))) v4i*)l, 0, 0);
#else
  *(float4*)l = *(const float4*)g;
#endif
}

__device__ inline void wait_async() {
#if HAVE_ASYNC_LDS
#if __has_builtin(__builtin_amdgcn_s_wait_asynccnt)
  __builtin_amdgcn_s_wait_asynccnt(0);
#else
  asm volatile("s_wait_asynccnt 0" ::: "memory");
#endif
#endif
}

// Load a 16x32 f16 fragment (A-layout; also B-layout when source is stored
// transposed, i.e. K contiguous per row).  Per CDNA5 ISA 7.12.2:
//   lane 0-15: row = lane,   VGPR v<4: K = half*8 + v*2,   v>=4: K = 16+...
// => per lane: 8 contiguous halfs at [row][half*8] and 8 at [row][16+half*8].
__device__ inline v16h load_frag16(const _Float16* base, int stride_h, int lane) {
  const int m  = lane & 15;
  const int hf = lane >> 4;
  union { v16h v; float4 q[2]; } u;
  u.q[0] = *(const float4*)(base + (size_t)m * stride_h + hf * 8);
  u.q[1] = *(const float4*)(base + (size_t)m * stride_h + 16 + hf * 8);
  return u.v;
}

// ---------------------------------------------------------------------------
// fp32 -> f16 conversion, 4 elems/thread (sizes are multiples of 1024)
// ---------------------------------------------------------------------------
__global__ __launch_bounds__(256) void cvt_f16_kernel(const float* __restrict__ src,
                                                      _Float16* __restrict__ dst) {
  const int i = (blockIdx.x * 256 + threadIdx.x) * 4;
  float4 v = *(const float4*)(src + i);
  union { uint2 u; _Float16 h[4]; } p;
  p.h[0] = (_Float16)v.x; p.h[1] = (_Float16)v.y;
  p.h[2] = (_Float16)v.z; p.h[3] = (_Float16)v.w;
  *(uint2*)(dst + i) = p.u;
}

// ---------------------------------------------------------------------------
// GEMM: C = X @ W + bias, X:[4096,768] f16, W:[768,768] f16.
// Block tile 128(M) x 128(N), K-step 32, double-buffered LDS, async X copy,
// register-prefetched transposed W tile.  8 waves; wave = 16 rows x 128 cols.
// MODE 0: scatter to Q/K/V [B,H,L,64] f16.   MODE 1: += residual, fp32 out.
// ---------------------------------------------------------------------------
template <int MODE>
__global__ __launch_bounds__(256) void gemm_kernel(const _Float16* __restrict__ X,
                                                   const _Float16* __restrict__ W,
                                                   const float* __restrict__ bias,
                                                   const float* __restrict__ resid,
                                                   _Float16* __restrict__ Yh,
                                                   float* __restrict__ Yf) {
  __shared__ _Float16 xt[2][128 * 32];   // 2 x 8 KB
  __shared__ _Float16 wt[2][128 * 32];   // 2 x 8 KB, transposed: wt[n][k]
  const int mtile = blockIdx.x;          // 0..31
  const int ntile = blockIdx.y;          // 0..5
  const int t = threadIdx.x, wave = t >> 5, lane = t & 31;

  // X tile 128x32: 512 16B segments; async copy, natural layout.
  auto loadX = [&](int kb, int buf) {
    #pragma unroll
    for (int p = t; p < 512; p += 256) {
      const int row = p >> 2, c4 = p & 3;
      async_copy16(X + ((size_t)(mtile * 128 + row)) * 768 + kb * 32 + c4 * 8,
                   &xt[buf][row * 32 + c4 * 8]);
    }
  };
  // W tile 32(k)x128(n): prefetch to regs, store transposed (K contiguous).
  auto loadW = [&](int kb, float4 wreg[2]) {
    #pragma unroll
    for (int i = 0; i < 2; ++i) {
      const int p = t + i * 256;                      // 512 segments
      wreg[i] = *(const float4*)(W + ((size_t)(kb * 32 + (p >> 4))) * 768 +
                                 ntile * 128 + (p & 15) * 8);
    }
  };
  auto storeW = [&](int buf, const float4 wreg[2]) {
    #pragma unroll
    for (int i = 0; i < 2; ++i) {
      const int p = t + i * 256;
      const int k = p >> 4, nb = (p & 15) * 8;
      union { float4 f; _Float16 h[8]; } u; u.f = wreg[i];
      #pragma unroll
      for (int j = 0; j < 8; ++j) wt[buf][(nb + j) * 32 + k] = u.h[j];
    }
  };

  float4 wreg[2];
  loadX(0, 0);
  loadW(0, wreg);
  storeW(0, wreg);
  wait_async();
  __syncthreads();

  v8f acc[8] = {};
  int cur = 0;
  for (int kb = 0; kb < 24; ++kb) {
    const int nxt = cur ^ 1;
    if (kb + 1 < 24) { loadX(kb + 1, nxt); loadW(kb + 1, wreg); }
    const v16h a = load_frag16(xt[cur] + wave * 16 * 32, 32, lane);
    #pragma unroll
    for (int t2 = 0; t2 < 8; ++t2) {
      const v16h b = load_frag16(wt[cur] + t2 * 16 * 32, 32, lane);
      acc[t2] = WMMA_F16(a, b, acc[t2]);
    }
    if (kb + 1 < 24) storeW(nxt, wreg);
    wait_async();
    __syncthreads();
    cur = nxt;
  }

  const int n_lo = lane & 15, hf = lane >> 4;
  #pragma unroll
  for (int t2 = 0; t2 < 8; ++t2) {
    const int n = ntile * 128 + t2 * 16 + n_lo;
    const float bv = bias[n];
    #pragma unroll
    for (int r = 0; r < 8; ++r) {
      const int m = mtile * 128 + wave * 16 + r + 8 * hf;
      if (MODE == 0) {
        const int b = m >> 11, l = m & 2047;
        const int h = n >> 6, d = n & 63;
        Yh[(((size_t)(b * 12 + h)) * 2048 + l) * 64 + d] = (_Float16)(acc[t2][r] + bv);
      } else {
        const size_t idx = (size_t)m * 768 + n;
        Yf[idx] = acc[t2][r] + bv + resid[idx];
      }
    }
  }
}

// ---------------------------------------------------------------------------
// Flash attention: grid (24 bh, 16 q-blocks of 128).  Wave = 16 query rows.
// Double-buffered K/V^T tiles; K via async copy, V transposed through regs.
// ---------------------------------------------------------------------------
__global__ __launch_bounds__(256) void attn_kernel(const _Float16* __restrict__ Qh,
                                                   const _Float16* __restrict__ Kh,
                                                   const _Float16* __restrict__ Vh,
                                                   const int* __restrict__ amask,
                                                   _Float16* __restrict__ Ch) {
  __shared__ _Float16 klds[2][64 * 64];   // K tile, natural [key][d]    2x8 KB
  __shared__ _Float16 vtlds[2][64 * 64];  // V tile transposed [d][key]  2x8 KB
  __shared__ float    mlds[2][64];        // additive key mask
  __shared__ _Float16 plds[8 * 16 * 64];  // per-wave P staging          16 KB

  const int bh = blockIdx.x;              // 0..23
  const int qb = blockIdx.y;              // 0..15
  const int b  = bh / 12, h = bh % 12;
  const int t = threadIdx.x, wave = t >> 5, lane = t & 31;
  const int n_lo = lane & 15, hf = lane >> 4;

  // Q fragments stay in registers: 16 rows x 64 d
  const _Float16* qbase = Qh + (((size_t)bh * 2048) + qb * 128 + wave * 16) * 64;
  const v16h qa0 = load_frag16(qbase, 64, lane);
  const v16h qa1 = load_frag16(qbase + 32, 64, lane);

  auto loadK = [&](int j, int buf) {
    const _Float16* kb_ = Kh + (((size_t)bh * 2048) + j * 64) * 64;
    #pragma unroll
    for (int p = t; p < 512; p += 256) async_copy16(kb_ + p * 8, &klds[buf][p * 8]);
  };
  auto loadV = [&](int j, unsigned int vreg[8]) {
    const unsigned int* vb_ = (const unsigned int*)(Vh + (((size_t)bh * 2048) + j * 64) * 64);
    #pragma unroll
    for (int i = 0; i < 8; ++i) vreg[i] = vb_[t * 8 + i];
  };
  auto storeV = [&](int buf, const unsigned int vreg[8]) {
    #pragma unroll
    for (int i = 0; i < 8; ++i) {
      const int p = t * 8 + i;                // 2048 half-pairs
      const int l = p >> 5, d2 = (p & 31) * 2;
      union { unsigned int u; _Float16 hh[2]; } cv; cv.u = vreg[i];
      vtlds[buf][d2 * 64 + l]       = cv.hh[0];
      vtlds[buf][(d2 + 1) * 64 + l] = cv.hh[1];
    }
  };
  auto loadM = [&](int j, int buf) {
    if (t < 64) mlds[buf][t] = amask[b * 2048 + j * 64 + t] ? 0.f : -1e30f;
  };

  float mstate[8], lstate[8];
  v8f o[4] = {};
  #pragma unroll
  for (int r = 0; r < 8; ++r) { mstate[r] = -1e30f; lstate[r] = 0.f; }

  unsigned int vreg[8];
  loadK(0, 0); loadV(0, vreg); storeV(0, vreg); loadM(0, 0);
  wait_async();
  __syncthreads();

  int cur = 0;
  for (int j = 0; j < 32; ++j) {
    const int nxt = cur ^ 1;
    if (j + 1 < 32) { loadK(j + 1, nxt); loadV(j + 1, vreg); loadM(j + 1, nxt); }

    // S = (Q @ K^T) * 0.125 + mask
    v8f s[4];
    #pragma unroll
    for (int t2 = 0; t2 < 4; ++t2) {
      v8f z = {};
      z = WMMA_F16(qa0, load_frag16(klds[cur] + t2 * 16 * 64, 64, lane), z);
      z = WMMA_F16(qa1, load_frag16(klds[cur] + t2 * 16 * 64 + 32, 64, lane), z);
      s[t2] = z;
    }

    // online softmax per query row (row = r + 8*hf, shared by a 16-lane half)
    #pragma unroll
    for (int r = 0; r < 8; ++r) {
      float v0 = s[0][r] * 0.125f + mlds[cur][0  + n_lo];
      float v1 = s[1][r] * 0.125f + mlds[cur][16 + n_lo];
      float v2 = s[2][r] * 0.125f + mlds[cur][32 + n_lo];
      float v3 = s[3][r] * 0.125f + mlds[cur][48 + n_lo];
      float rmax = fmaxf(fmaxf(v0, v1), fmaxf(v2, v3));
      #pragma unroll
      for (int x = 8; x >= 1; x >>= 1) rmax = fmaxf(rmax, __shfl_xor(rmax, x, 32));
      const float newm  = fmaxf(mstate[r], rmax);
      const float alpha = __expf(mstate[r] - newm);
      const float e0 = (v0 > -1e29f) ? __expf(v0 - newm) : 0.f;
      const float e1 = (v1 > -1e29f) ? __expf(v1 - newm) : 0.f;
      const float e2 = (v2 > -1e29f) ? __expf(v2 - newm) : 0.f;
      const float e3 = (v3 > -1e29f) ? __expf(v3 - newm) : 0.f;
      float rsum = e0 + e1 + e2 + e3;
      #pragma unroll
      for (int x = 8; x >= 1; x >>= 1) rsum += __shfl_xor(rsum, x, 32);
      lstate[r] = lstate[r] * alpha + rsum;
      mstate[r] = newm;
      #pragma unroll
      for (int t2 = 0; t2 < 4; ++t2) o[t2][r] *= alpha;
      _Float16* pr = plds + wave * 1024 + (r + 8 * hf) * 64 + n_lo;
      pr[0]  = (_Float16)e0;  pr[16] = (_Float16)e1;
      pr[32] = (_Float16)e2;  pr[48] = (_Float16)e3;
    }

    // intra-wave LDS RAW (cross-lane): wait for the P stores, then O += P @ V
    asm volatile("s_wait_dscnt 0" ::: "memory");
    const _Float16* pw = plds + wave * 1024;
    const v16h pa0 = load_frag16(pw, 64, lane);
    const v16h pa1 = load_frag16(pw + 32, 64, lane);
    #pragma unroll
    for (int t2 = 0; t2 < 4; ++t2) {
      o[t2] = WMMA_F16(pa0, load_frag16(vtlds[cur] + t2 * 16 * 64, 64, lane), o[t2]);
      o[t2] = WMMA_F16(pa1, load_frag16(vtlds[cur] + t2 * 16 * 64 + 32, 64, lane), o[t2]);
    }

    if (j + 1 < 32) storeV(nxt, vreg);
    wait_async();
    __syncthreads();
    cur = nxt;
  }

  // epilogue: query-mask * (O / l) -> combined [B,L,768] f16
  #pragma unroll
  for (int r = 0; r < 8; ++r) {
    const int qrow = qb * 128 + wave * 16 + r + 8 * hf;
    const int qm = amask[b * 2048 + qrow];
    const float scl = (qm != 0 && lstate[r] > 0.f) ? (1.f / lstate[r]) : 0.f;
    #pragma unroll
    for (int t2 = 0; t2 < 4; ++t2) {
      const int d = t2 * 16 + n_lo;
      Ch[((size_t)(b * 2048 + qrow)) * 768 + h * 64 + d] = (_Float16)(o[t2][r] * scl);
    }
  }
}

// ---------------------------------------------------------------------------
// LayerNorm over the SEQUENCE axis (axis=1), ddof=1, per (batch, channel).
// grid (2, 12); block 256: 64 channels x 4 row-segments, channel-coalesced.
// ---------------------------------------------------------------------------
__global__ __launch_bounds__(256) void ln_kernel(const float* __restrict__ Xr,
                                                 const float* __restrict__ gamma,
                                                 const float* __restrict__ beta,
                                                 float* __restrict__ out) {
  __shared__ float ssum[256], ssq[256];
  __shared__ float smean[64], sinv[64];
  const int b = blockIdx.x;
  const int c = blockIdx.y * 64 + (threadIdx.x & 63);
  const int seg = threadIdx.x >> 6;
  float sum = 0.f, sq = 0.f;
  for (int l = seg; l < 2048; l += 4) {
    const float v = Xr[((size_t)b * 2048 + l) * 768 + c];
    sum += v; sq += v * v;
  }
  ssum[threadIdx.x] = sum; ssq[threadIdx.x] = sq;
  __syncthreads();
  if (seg == 0) {
    const int i = threadIdx.x;
    const float S = ssum[i] + ssum[i + 64] + ssum[i + 128] + ssum[i + 192];
    const float Q = ssq[i] + ssq[i + 64] + ssq[i + 128] + ssq[i + 192];
    const float mean = S * (1.f / 2048.f);
    const float var = (Q - 2048.f * mean * mean) * (1.f / 2047.f);
    smean[i] = mean;
    sinv[i] = rsqrtf(fmaxf(var, 1e-20f));
  }
  __syncthreads();
  const float mean = smean[threadIdx.x & 63], inv = sinv[threadIdx.x & 63];
  const float g = gamma[c], bb = beta[c];
  for (int l = seg; l < 2048; l += 4) {
    const size_t idx = ((size_t)b * 2048 + l) * 768 + c;
    out[idx] = (Xr[idx] - mean) * inv * g + bb;
  }
}

// ---------------------------------------------------------------------------
extern "C" void kernel_launch(void* const* d_in, const int* in_sizes, int n_in,
                              void* d_out, int out_size, void* d_ws, size_t ws_size,
                              hipStream_t stream) {
  const float* inputs = (const float*)d_in[0];   // [2,2048,768]
  const int*   amask  = (const int*)d_in[1];     // [2,2048]
  const float* wq_w = (const float*)d_in[2];
  const float* wq_b = (const float*)d_in[3];
  const float* wk_w = (const float*)d_in[4];
  const float* wk_b = (const float*)d_in[5];
  const float* wv_w = (const float*)d_in[6];
  const float* wv_b = (const float*)d_in[7];
  const float* wo_w = (const float*)d_in[8];
  const float* wo_b = (const float*)d_in[9];
  const float* gamma = (const float*)d_in[10];
  const float* beta  = (const float*)d_in[11];
  float* out = (float*)d_out;

  // workspace carve-up (~48.8 MB), 256B aligned
  char* ws = (char*)d_ws;
  size_t off = 0;
  auto carve = [&](size_t bytes) { char* p = ws + off; off += (bytes + 255) & ~(size_t)255; return p; };
  const size_t NX = (size_t)4096 * 768;          // activations
  const size_t NW = (size_t)768 * 768;           // weights
  _Float16* xh  = (_Float16*)carve(NX * 2);
  _Float16* wqh = (_Float16*)carve(NW * 2);
  _Float16* wkh = (_Float16*)carve(NW * 2);
  _Float16* wvh = (_Float16*)carve(NW * 2);
  _Float16* woh = (_Float16*)carve(NW * 2);
  _Float16* qhh = (_Float16*)carve(NX * 2);      // [B,H,L,64]
  _Float16* khh = (_Float16*)carve(NX * 2);
  _Float16* vhh = (_Float16*)carve(NX * 2);
  _Float16* chh = (_Float16*)carve(NX * 2);      // combined [B,L,768]
  float*    xr  = (float*)carve(NX * 4);         // residual sum, fp32
  (void)ws_size; (void)in_sizes; (void)n_in; (void)out_size;

  // 1) fp32 -> f16 conversions
  cvt_f16_kernel<<<NX / 1024, 256, 0, stream>>>(inputs, xh);
  cvt_f16_kernel<<<NW / 1024, 256, 0, stream>>>(wq_w, wqh);
  cvt_f16_kernel<<<NW / 1024, 256, 0, stream>>>(wk_w, wkh);
  cvt_f16_kernel<<<NW / 1024, 256, 0, stream>>>(wv_w, wvh);
  cvt_f16_kernel<<<NW / 1024, 256, 0, stream>>>(wo_w, woh);

  // 2) Q/K/V projections (128x128 tiles)
  dim3 ggrid(32, 6);
  gemm_kernel<0><<<ggrid, 256, 0, stream>>>(xh, wqh, wq_b, nullptr, qhh, nullptr);
  gemm_kernel<0><<<ggrid, 256, 0, stream>>>(xh, wkh, wk_b, nullptr, khh, nullptr);
  gemm_kernel<0><<<ggrid, 256, 0, stream>>>(xh, wvh, wv_b, nullptr, vhh, nullptr);

  // 3) fused attention
  attn_kernel<<<dim3(24, 16), 256, 0, stream>>>(qhh, khh, vhh, amask, chh);

  // 4) output projection + residual
  gemm_kernel<1><<<ggrid, 256, 0, stream>>>(chh, woh, wo_b, inputs, nullptr, xr);

  // 5) layernorm over sequence axis
  ln_kernel<<<dim3(2, 12), 256, 0, stream>>>(xr, gamma, beta, out);
}
